// SimpleLSTM_5093831213205
// MI455X (gfx1250) — matrix operand order repaired and checked by reading, not compiled, not run on hardware
//
#include <hip/hip_runtime.h>
#include <hip/hip_bf16.h>

typedef unsigned short u16;
typedef __attribute__((ext_vector_type(16))) __bf16 v16bf;
typedef __attribute__((ext_vector_type(8)))  float  v8f;

#define Bc 128
#define Dc 256
#define Tc 512
#define Hc 1024
#define G4 4096   /* 4*H */

// ---------- helpers ----------

__device__ __forceinline__ u16 f2bf(float f) {
    unsigned int u = __float_as_uint(f);
    unsigned int r = u + 0x7FFFu + ((u >> 16) & 1u);   // round-to-nearest-even
    return (u16)(r >> 16);
}

struct F2 { uint4 a, b; };

// Build a 16-element bf16 fragment from two 16-byte chunks (global_load_b128 pairs)
__device__ __forceinline__ v16bf ld2(const u16* p0, const u16* p1) {
    F2 f;
    f.a = *reinterpret_cast<const uint4*>(p0);
    f.b = *reinterpret_cast<const uint4*>(p1);
    return __builtin_bit_cast(v16bf, f);
}

// A-matrix 16x32 bf16, wave32 layout (CDNA5 ISA):
//   lanes 0-15 : M=lane,    K = {0..7} in v0-3,  {16..23} in v4-7
//   lanes 16-31: M=lane-16, K = {8..15},         {24..31}
__device__ __forceinline__ v16bf load_a(const u16* row, int k0, int hi) {
    const u16* p = row + k0 + hi * 8;
    return ld2(p, p + 16);
}

// B-matrix 32x16 bf16: lane n<16 holds column n, K={0..7,16..23}; lane n+16 holds K={8..15,24..31}.
// B(k,n) = W[col0+n][k] (W row-major, inner dim = K) => two runs of 8 contiguous bf16 per lane, 16 apart.
__device__ __forceinline__ v16bf load_b(const u16* wrow, int k0, int hi) {
    const u16* p = wrow + k0 + hi * 8;
    return ld2(p, p + 16);
}

__device__ __forceinline__ v8f wmma_bf16(v16bf a, v16bf b, v8f c) {
    return __builtin_amdgcn_wmma_f32_16x16x32_bf16(
        /*neg_a=*/false, a, /*neg_b=*/false, b,
        /*c_mod=*/(short)0, c, /*reuse_a=*/false, /*reuse_b=*/false);
}

__device__ __forceinline__ float sigm(float x) {
    return 1.0f / (1.0f + __expf(-x));
}

// Software-pipelined K-loop: load k+32 fragments before the WMMAs for k.
// arow: A row pointer (k contiguous). wbase: B weight matrix, rows = output cols,
// LD = B row stride (== K extent here), NK = K extent, colm = col0 + (lane&15).
template<int LD, int NK>
__device__ __forceinline__ void gemm_piece(const u16* __restrict__ arow,
                                           const u16* __restrict__ wbase,
                                           int colm, int hi, v8f acc[4])
{
    const u16* w0 = wbase + (size_t)(colm +  0) * LD;
    const u16* w1 = wbase + (size_t)(colm + 16) * LD;
    const u16* w2 = wbase + (size_t)(colm + 32) * LD;
    const u16* w3 = wbase + (size_t)(colm + 48) * LD;

    v16bf a  = load_a(arow, 0, hi);
    v16bf b0 = load_b(w0, 0, hi);
    v16bf b1 = load_b(w1, 0, hi);
    v16bf b2 = load_b(w2, 0, hi);
    v16bf b3 = load_b(w3, 0, hi);

    for (int k0 = 0; k0 < NK - 32; k0 += 32) {
        const int kn = k0 + 32;
        // prefetch far ahead so L1 stays warm for the strip
        __builtin_prefetch(w0 + kn + 224, 0, 1);
        v16bf an  = load_a(arow, kn, hi);
        v16bf bn0 = load_b(w0, kn, hi);
        v16bf bn1 = load_b(w1, kn, hi);
        v16bf bn2 = load_b(w2, kn, hi);
        v16bf bn3 = load_b(w3, kn, hi);
        acc[0] = wmma_bf16(a, b0, acc[0]);
        acc[1] = wmma_bf16(a, b1, acc[1]);
        acc[2] = wmma_bf16(a, b2, acc[2]);
        acc[3] = wmma_bf16(a, b3, acc[3]);
        a = an; b0 = bn0; b1 = bn1; b2 = bn2; b3 = bn3;
    }
    acc[0] = wmma_bf16(a, b0, acc[0]);
    acc[1] = wmma_bf16(a, b1, acc[1]);
    acc[2] = wmma_bf16(a, b2, acc[2]);
    acc[3] = wmma_bf16(a, b3, acc[3]);
}

// ---------- prep kernels (run once per launch) ----------

__global__ __launch_bounds__(256) void prep_params(
    const float* __restrict__ Wih_f, const float* __restrict__ Whh_f,
    const float* __restrict__ bih,   const float* __restrict__ bhh,
    u16* __restrict__ Wih_b, u16* __restrict__ Whh_b, float* __restrict__ bb,
    u16* __restrict__ h0, float* __restrict__ c)
{
    size_t idx = (size_t)blockIdx.x * 256 + threadIdx.x;
    const size_t NIH = (size_t)G4 * Dc;
    const size_t NHH = (size_t)G4 * Hc;
    const size_t NB  = G4;
    const size_t NH  = (size_t)Bc * Hc;
    if (idx < NIH) {
        Wih_b[idx] = f2bf(Wih_f[idx]);
    } else if (idx < NIH + NHH) {
        size_t i = idx - NIH;
        Whh_b[i] = f2bf(Whh_f[i]);
    } else if (idx < NIH + NHH + NB) {
        size_t i = idx - NIH - NHH;
        bb[i] = bih[i] + bhh[i];
    } else if (idx < NIH + NHH + NB + NH) {
        h0[idx - NIH - NHH - NB] = 0;
    } else if (idx < NIH + NHH + NB + 2 * NH) {
        c[idx - NIH - NHH - NB - NH] = 0.0f;
    }
}

// x [B, D, T] f32  ->  xT [T, B, D] bf16 (per-step contiguous A tiles)
__global__ __launch_bounds__(256) void prep_x(
    const float* __restrict__ x, u16* __restrict__ xT)
{
    size_t idx = (size_t)blockIdx.x * 256 + threadIdx.x;
    if (idx >= (size_t)Bc * Dc * Tc) return;
    int t  = (int)(idx % Tc);
    size_t bd = idx / Tc;
    int d  = (int)(bd % Dc);
    int b  = (int)(bd / Dc);
    xT[((size_t)t * Bc + b) * Dc + d] = f2bf(x[idx]);
}

// ---------- per-step gates GEMM (pure WMMA, pipelined) ----------
// gates[128,4096] = xT[t](128x256) @ Wih^T + hprev(128x1024) @ Whh^T   (bf16 in, f32 acc)
// Mapping for WGP$ locality on B: blockIdx.x = 64-column strip (64 blocks),
// wave-in-block = M-tile (8 waves). All 8 consumers of a W row share one WGP's L1.
__global__ __launch_bounds__(256) void lstm_step_gemm(
    const u16* __restrict__ xT, const u16* __restrict__ Wih,
    const u16* __restrict__ Whh, const u16* __restrict__ hprev,
    float* __restrict__ gates, int t)
{
    const int lane  = threadIdx.x & 31;
    const int mtile = threadIdx.x >> 5;        // [0,8) : M-tile of this wave
    const int col0  = blockIdx.x * 64;         // [0,4096) : N strip of this block
    const int m  = lane & 15;
    const int hi = lane >> 4;
    const int colm = col0 + m;

    v8f acc[4];
    acc[0] = {}; acc[1] = {}; acc[2] = {}; acc[3] = {};

    // Part 1: K over D (input projection)
    const u16* arow_x = xT + ((size_t)t * Bc + mtile * 16 + m) * Dc;
    gemm_piece<Dc, Dc>(arow_x, Wih, colm, hi, acc);

    // Part 2: K over H (recurrent projection)
    const u16* arow_h = hprev + (size_t)(mtile * 16 + m) * Hc;
    gemm_piece<Hc, Hc>(arow_h, Whh, colm, hi, acc);

    // Store: C/D layout — VGPR r: M = r + 8*hi, N = lane&15
    const int row0 = mtile * 16 + hi * 8;
#pragma unroll
    for (int j = 0; j < 4; ++j) {
        float* gp = gates + (size_t)row0 * G4 + col0 + j * 16 + m;
#pragma unroll
        for (int r = 0; r < 8; ++r)
            gp[(size_t)r * G4] = acc[j][r];
    }
}

// ---------- per-step pointwise: gates -> (c,h), y = h . W_out + b_out ----------
__global__ __launch_bounds__(256) void lstm_step_point(
    const float* __restrict__ gates, const float* __restrict__ bb,
    float* __restrict__ c, u16* __restrict__ hnext,
    const float* __restrict__ Wout, const float* __restrict__ bout,
    float* __restrict__ out, int t)
{
    const int b   = blockIdx.x;
    const int tid = threadIdx.x;
    const float* gr = gates + (size_t)b * G4;
    float acc = 0.0f;

#pragma unroll
    for (int jj = 0; jj < 4; ++jj) {
        int j = jj * 256 + tid;
        float gi = gr[j]          + bb[j];
        float gf = gr[Hc + j]     + bb[Hc + j];
        float gg = gr[2 * Hc + j] + bb[2 * Hc + j];
        float go = gr[3 * Hc + j] + bb[3 * Hc + j];
        float i_ = sigm(gi);
        float f_ = sigm(gf);
        float g_ = tanhf(gg);
        float o_ = sigm(go);
        size_t ci = (size_t)b * Hc + j;
        float cn = f_ * c[ci] + i_ * g_;
        c[ci] = cn;
        float h = o_ * tanhf(cn);
        hnext[ci] = f2bf(h);           // bf16 A-operand for next step's WMMA
        acc += h * Wout[j];
    }

    for (int off = 16; off > 0; off >>= 1)
        acc += __shfl_down(acc, off, 32);
    __shared__ float red[8];
    if ((tid & 31) == 0) red[tid >> 5] = acc;
    __syncthreads();
    if (tid == 0) {
        float s = 0.0f;
#pragma unroll
        for (int w = 0; w < 8; ++w) s += red[w];
        out[(size_t)b * Tc + t] = s + bout[0];   // out[b, t], F=1
    }
}

// ---------- launch ----------

extern "C" void kernel_launch(void* const* d_in, const int* in_sizes, int n_in,
                              void* d_out, int out_size, void* d_ws, size_t ws_size,
                              hipStream_t stream) {
    const float* x    = (const float*)d_in[0];
    const float* Wih  = (const float*)d_in[1];
    const float* Whh  = (const float*)d_in[2];
    const float* bih  = (const float*)d_in[3];
    const float* bhh  = (const float*)d_in[4];
    const float* Wout = (const float*)d_in[5];
    const float* bout = (const float*)d_in[6];
    float* out = (float*)d_out;

    char* w = (char*)d_ws;
    u16*  xT    = (u16*)w;  w += (size_t)Tc * Bc * Dc * sizeof(u16);   // 32 MB
    u16*  Wihb  = (u16*)w;  w += (size_t)G4 * Dc * sizeof(u16);        //  2 MB
    u16*  Whhb  = (u16*)w;  w += (size_t)G4 * Hc * sizeof(u16);        //  8 MB
    float* bb   = (float*)w; w += (size_t)G4 * sizeof(float);
    float* gates= (float*)w; w += (size_t)Bc * G4 * sizeof(float);     //  2 MB
    u16*  h0    = (u16*)w;  w += (size_t)Bc * Hc * sizeof(u16);
    u16*  h1    = (u16*)w;  w += (size_t)Bc * Hc * sizeof(u16);
    float* c    = (float*)w; w += (size_t)Bc * Hc * sizeof(float);

    {
        size_t total = (size_t)G4 * Dc + (size_t)G4 * Hc + G4 + 2 * (size_t)Bc * Hc;
        int grid = (int)((total + 255) / 256);
        prep_params<<<grid, 256, 0, stream>>>(Wih, Whh, bih, bhh, Wihb, Whhb, bb, h0, c);
    }
    {
        size_t total = (size_t)Bc * Dc * Tc;
        int grid = (int)((total + 255) / 256);
        prep_x<<<grid, 256, 0, stream>>>(x, xT);
    }

    u16* hbuf[2] = { h0, h1 };
    for (int t = 0; t < Tc; ++t) {
        lstm_step_gemm<<<64, 256, 0, stream>>>(xT, Wihb, Whhb, hbuf[t & 1], gates, t);
        lstm_step_point<<<128, 256, 0, stream>>>(gates, bb, c, hbuf[(t + 1) & 1],
                                                 Wout, bout, out, t);
    }
}
